// SelfAttentionSeq_63144609186429
// MI455X (gfx1250) — matrix operand order repaired and checked by reading, not compiled, not run on hardware
//
#include <hip/hip_runtime.h>
#include <hip/hip_bf16.h>
#include <math.h>

// SelfAttentionSeq (additive attention pooling) for MI455X / gfx1250.
// e = tanh(h @ a) @ b ; softmax over S with mask ; out = attn @ h
// B=64, S=4096, DIM=256, DA=64.

typedef __attribute__((ext_vector_type(16))) __bf16 v16bf;
typedef __attribute__((ext_vector_type(8)))  float  v8f;

static constexpr int BB  = 64;
static constexpr int SS  = 4096;
static constexpr int DIM = 256;
static constexpr int DA  = 64;
static constexpr long NROWS = (long)BB * SS;      // 262144 GEMM rows
static constexpr float NEGBIG = -1e30f;

// ---------------------------------------------------------------------------
// Kernel 1: logits.  Each wave computes a 16-row tile of the (NROWS x 256) x
// (256 x 64) GEMM with v_wmma_f32_16x16x32_bf16, applies tanh, dots with b,
// reduces across lanes, applies mask, writes e[row].
// Block = 256 threads = 8 waves = 128 rows.  Grid = 2048 blocks.
// ---------------------------------------------------------------------------
__global__ __launch_bounds__(256) void k_logits(
    const float* __restrict__ h,          // (B*S, 256) row-major
    const unsigned char* __restrict__ mask, // (B*S) bool
    const float* __restrict__ aw,         // (256, 64) row-major
    const float* __restrict__ bw,         // (64)
    float* __restrict__ e_out)            // (B*S)
{
    // LDS: weight matrix pre-swizzled into WMMA B-fragment layout.
    // Fragment index [kc(8)][t(4)][lane(32)][elem(16)] (bf16) = 32 KB.
    __shared__ __attribute__((aligned(32))) __bf16 sa[8 * 4 * 32 * 16];
    __shared__ float sb[DA];

    const int tid = threadIdx.x;

    // Stage a -> bf16 B-fragments.  B layout (16-bit, 32x16 tile):
    // lane L holds column n = L&15; lanes 0-15 hold K = kb+{0..7,16..23},
    // lanes 16-31 hold K = kb+{8..15,24..31}.
    for (int i = tid; i < 8 * 4 * 32 * 16; i += 256) {
        const int j  = i & 15;
        const int ln = (i >> 4) & 31;
        const int t  = (i >> 9) & 3;
        const int kc = i >> 11;
        const int k  = kc * 32 + ((ln >> 4) << 3) + (j & 7) + ((j >> 3) << 4);
        const int c  = t * 16 + (ln & 15);
        sa[i] = (__bf16)aw[k * DA + c];
    }
    if (tid < DA) sb[tid] = bw[tid];
    __syncthreads();

    const int wave  = tid >> 5;
    const int lane  = tid & 31;
    const int mrow  = lane & 15;       // row within 16-row tile (A layout)
    const int khalf = lane >> 4;       // 0: K {0..7,16..23}; 1: K {8..15,24..31}

    const long m0 = ((long)blockIdx.x * 8 + wave) * 16;
    const float* __restrict__ hrow = h + (m0 + mrow) * DIM;

    v8f acc[4] = {};

#pragma unroll
    for (int kc = 0; kc < 8; ++kc) {
        const int kb = kc * 32;
        // A fragment: 16-bit A 16x32 layout.  This lane needs
        // K = kb+koff+{0..7} and kb+koff+16+{0..7}, koff = khalf*8.
        const float* hp = hrow + kb + khalf * 8;
        float ta[16];
        *(float4*)(ta + 0)  = *(const float4*)(hp + 0);
        *(float4*)(ta + 4)  = *(const float4*)(hp + 4);
        *(float4*)(ta + 8)  = *(const float4*)(hp + 16);
        *(float4*)(ta + 12) = *(const float4*)(hp + 20);
        v16bf af;
#pragma unroll
        for (int j = 0; j < 16; ++j) af[j] = (__bf16)ta[j];

#pragma unroll
        for (int t = 0; t < 4; ++t) {
            const v16bf bf = *(const v16bf*)&sa[((kc * 4 + t) * 32 + lane) * 16];
            acc[t] = __builtin_amdgcn_wmma_f32_16x16x32_bf16(
                /*neg_a=*/false, af, /*neg_b=*/false, bf,
                /*c_mod=*/(short)0, acc[t], /*reuse_a=*/false, /*reuse_b=*/false);
        }
    }

    // Epilogue: C layout -> VGPR r holds row (m0 + r + 8*khalf), lane&15 = n.
    // e contribution of this lane: sum_t tanh(acc[t][r]) * b[t*16 + n].
    float er[8] = {0.f, 0.f, 0.f, 0.f, 0.f, 0.f, 0.f, 0.f};
#pragma unroll
    for (int t = 0; t < 4; ++t) {
        const float bwv = sb[t * 16 + (lane & 15)];
#pragma unroll
        for (int r = 0; r < 8; ++r)
            er[r] += tanhf(acc[t][r]) * bwv;
    }
    // Reduce over the 16 N-lanes within each half-wave.
#pragma unroll
    for (int m = 1; m <= 8; m <<= 1) {
#pragma unroll
        for (int r = 0; r < 8; ++r)
            er[r] += __shfl_xor(er[r], m, 32);
    }
    if ((lane & 15) == 0) {
        const long rowbase = m0 + (khalf << 3);
#pragma unroll
        for (int r = 0; r < 8; ++r) {
            const long row = rowbase + r;
            float ev = er[r];
            if (mask[row]) ev += NEGBIG;   // True == padding
            e_out[row] = ev;
        }
    }
}

// ---------------------------------------------------------------------------
// Kernel 2: per-batch softmax over S=4096.  One block per batch.
// ---------------------------------------------------------------------------
__global__ __launch_bounds__(256) void k_softmax(
    const float* __restrict__ e, float* __restrict__ w)
{
    __shared__ float red[256];
    const int b = blockIdx.x, tid = threadIdx.x;
    const float* eb = e + (size_t)b * SS;

    float lm = -3.4e38f;
    for (int s = tid; s < SS; s += 256) lm = fmaxf(lm, eb[s]);
    red[tid] = lm; __syncthreads();
#pragma unroll
    for (int o = 128; o > 0; o >>= 1) {
        if (tid < o) red[tid] = fmaxf(red[tid], red[tid + o]);
        __syncthreads();
    }
    const float gm = red[0]; __syncthreads();

    float ls = 0.f;
    for (int s = tid; s < SS; s += 256) ls += __expf(eb[s] - gm);
    red[tid] = ls; __syncthreads();
#pragma unroll
    for (int o = 128; o > 0; o >>= 1) {
        if (tid < o) red[tid] += red[tid + o];
        __syncthreads();
    }
    const float inv = 1.0f / red[0];

    for (int s = tid; s < SS; s += 256)
        w[(size_t)b * SS + s] = __expf(eb[s] - gm) * inv;
}

// ---------------------------------------------------------------------------
// Kernel 3: weighted sum partials.  Grid (16 S-chunks, 64 batches), 256 thr.
// Thread = one d column (DIM==256) -> fully coalesced h reads.
// ---------------------------------------------------------------------------
__global__ __launch_bounds__(256) void k_wsum(
    const float* __restrict__ h, const float* __restrict__ w,
    float* __restrict__ part)             // [16][64][256]
{
    __shared__ float sw[256];
    const int c = blockIdx.x, b = blockIdx.y, d = threadIdx.x;
    const int s0 = c * 256;
    sw[d] = w[(size_t)b * SS + s0 + d];
    __syncthreads();

    const float* __restrict__ hb = h + ((size_t)b * SS + s0) * DIM + d;
    float acc = 0.f;
#pragma unroll 4
    for (int sl = 0; sl < 256; ++sl)
        acc += sw[sl] * hb[(size_t)sl * DIM];

    part[((size_t)c * BB + b) * DIM + d] = acc;
}

// ---------------------------------------------------------------------------
// Kernel 4: fixed-order reduction of the 16 partials (deterministic fp).
// ---------------------------------------------------------------------------
__global__ __launch_bounds__(256) void k_reduce(
    const float* __restrict__ part, float* __restrict__ out)
{
    const int b = blockIdx.x, d = threadIdx.x;
    float acc = 0.f;
#pragma unroll
    for (int c = 0; c < 16; ++c)
        acc += part[((size_t)c * BB + b) * DIM + d];
    out[(size_t)b * DIM + d] = acc;
}

// ---------------------------------------------------------------------------
extern "C" void kernel_launch(void* const* d_in, const int* in_sizes, int n_in,
                              void* d_out, int out_size, void* d_ws, size_t ws_size,
                              hipStream_t stream) {
    (void)in_sizes; (void)n_in; (void)out_size; (void)ws_size;

    const float*         h    = (const float*)d_in[0];         // (64,4096,256) f32
    const unsigned char* mask = (const unsigned char*)d_in[1]; // (64,4096) bool
    const float*         aw   = (const float*)d_in[2];         // (256,64) f32
    const float*         bw   = (const float*)d_in[3];         // (64,1) f32
    float*               out  = (float*)d_out;                 // (64,256) f32

    float* wsf = (float*)d_ws;                 // needs 3 * 1 MB
    float* e   = wsf;                          // B*S logits
    float* w   = wsf + NROWS;                  // B*S attn weights
    float* p   = wsf + 2 * NROWS;              // 16*64*256 partials

    k_logits <<<dim3(2048),       256, 0, stream>>>(h, mask, aw, bw, e);
    k_softmax<<<dim3(BB),         256, 0, stream>>>(e, w);
    k_wsum   <<<dim3(16, BB),     256, 0, stream>>>(h, w, p);
    k_reduce <<<dim3(BB),         256, 0, stream>>>(p, out);
}
